// LinkEncoder_1606317769408
// MI455X (gfx1250) — matrix operand order repaired and checked by reading, not compiled, not run on hardware
//
#include <hip/hip_runtime.h>
#include <math.h>

#define SZ    20
#define HID   172
#define TD    84
#define DIMS  256
#define ITEMS 4
#define ROWS  (ITEMS * SZ)   // 80
#define LDF   264            // padded row stride (elements) for Xf (f32) and Xbf (bf16)
#define NTILES (DIMS / 16)   // 16
#define MTILES (ROWS / 16)   // 5
#define KTILES (DIMS / 32)   // 8

typedef __attribute__((ext_vector_type(16))) __bf16 v16bf;
typedef __attribute__((ext_vector_type(8)))  float  v8f;
typedef __attribute__((ext_vector_type(8)))  int    v8i;
typedef __attribute__((ext_vector_type(4)))  int    v4i;

__device__ __forceinline__ unsigned short f2bf(float f) {
    unsigned u = __builtin_bit_cast(unsigned, f);
    unsigned r = (u + 0x7FFFu + ((u >> 16) & 1u)) >> 16;   // round-to-nearest-even
    return (unsigned short)r;
}

__device__ __forceinline__ float gelu_exact(float x) {
    return 0.5f * x * (1.0f + erff(x * 0.70710678118654752f));
}

// ---------------------------------------------------------------------------
// Prep: convert both weight matrices to bf16 in workspace (L2-resident reuse)
// ---------------------------------------------------------------------------
__global__ void convert_weights_kernel(const float* __restrict__ tokw,
                                       const float* __restrict__ chw,
                                       unsigned short* __restrict__ tokbf,
                                       unsigned short* __restrict__ chbf,
                                       int n) {
    int i = blockIdx.x * blockDim.x + threadIdx.x;
    if (i < n) {
        tokbf[i] = f2bf(tokw[i]);
        chbf[i]  = f2bf(chw[i]);
    }
}

// ---------------------------------------------------------------------------
// One mixer block: LN -> (bf16) -> WMMA GEMM vs W^T -> bias -> GELU -> +=Xf
// ---------------------------------------------------------------------------
__device__ __forceinline__ void mixer_block(float* Xf, unsigned short* Xbf,
                                            float* muv, float* rsv,
                                            const float* __restrict__ ln_g,
                                            const float* __restrict__ ln_b,
                                            const unsigned short* __restrict__ Wbf,
                                            const float* __restrict__ bias,
                                            int wave, int lane, int tid) {
    // --- LayerNorm statistics: one wave per row (strided), butterfly reduce ---
    for (int r = wave; r < ROWS; r += 8) {
        float s = 0.f, s2 = 0.f;
        #pragma unroll
        for (int i = 0; i < DIMS / 32; ++i) {
            float v = Xf[r * LDF + lane + 32 * i];
            s += v; s2 += v * v;
        }
        #pragma unroll
        for (int off = 16; off >= 1; off >>= 1) {
            s  += __shfl_xor(s,  off, 32);
            s2 += __shfl_xor(s2, off, 32);
        }
        if (lane == 0) {
            float mu  = s * (1.0f / DIMS);
            float var = s2 * (1.0f / DIMS) - mu * mu;
            muv[r] = mu;
            rsv[r] = rsqrtf(var + 1e-5f);
        }
    }
    __syncthreads();

    // --- Normalize + affine + convert to bf16 A-operand staging ---
    for (int idx = tid; idx < ROWS * DIMS; idx += 256) {
        int r = idx >> 8, c = idx & 255;
        float v = (Xf[r * LDF + c] - muv[r]) * rsv[r] * ln_g[c] + ln_b[c];
        Xbf[r * LDF + c] = f2bf(v);
    }
    __syncthreads();

    // --- WMMA GEMM: y[m,n] = sum_k Xn[m,k] * W[n,k] ---
    const int h  = lane >> 4;   // half-wave select
    const int lm = lane & 15;
    for (int ntc = 0; ntc < 2; ++ntc) {
        const int nt   = wave + ntc * 8;          // this wave's N-tile
        const int ncol = nt * 16 + lm;
        v8f acc[MTILES];
        #pragma unroll
        for (int mt = 0; mt < MTILES; ++mt)
            acc[mt] = (v8f){0.f, 0.f, 0.f, 0.f, 0.f, 0.f, 0.f, 0.f};

        for (int kt = 0; kt < KTILES; ++kt) {
            // B fragment (32x16, bf16): lane holds W[ncol][kt*32+16h .. +15],
            // 32 contiguous bytes -> vector load (L2-resident weights).
            v8i bp8 = *(const v8i*)(Wbf + (size_t)ncol * DIMS + kt * 32 + 16 * h);
            v16bf bfrag = __builtin_bit_cast(v16bf, bp8);

            #pragma unroll
            for (int mt = 0; mt < MTILES; ++mt) {
                const int m = mt * 16 + lm;
                // A fragment (16x32, bf16): two contiguous 16B runs per lane:
                // elements [kt*32+16h .. +7] and [kt*32+16h+8 .. +7]
                const v4i* aprow = (const v4i*)(Xbf + (size_t)m * LDF);
                v4i lo = aprow[kt * 4 + 2 * h];
                v4i hi = aprow[kt * 4 + 2 * h + 1];
                v8i ap8 = {lo.x, lo.y, lo.z, lo.w, hi.x, hi.y, hi.z, hi.w};
                v16bf afrag = __builtin_bit_cast(v16bf, ap8);
                acc[mt] = __builtin_amdgcn_wmma_f32_16x16x32_bf16(
                    false, afrag, false, bfrag, (short)0, acc[mt], false, false);
            }
        }

        // --- Epilogue: bias + exact GELU + residual accumulate into Xf ---
        const float bn = bias[ncol];
        #pragma unroll
        for (int mt = 0; mt < MTILES; ++mt) {
            #pragma unroll
            for (int j = 0; j < 8; ++j) {
                int row = mt * 16 + j + 8 * h;       // C/D layout: M = j + 8*(l/16)
                float y = acc[mt][j] + bn;
                Xf[row * LDF + ncol] += gelu_exact(y);
            }
        }
    }
    __syncthreads();
}

// ---------------------------------------------------------------------------
// Fused kernel: gather + temporal enc + 2 mixer blocks + slot-mean
// ---------------------------------------------------------------------------
__global__ __launch_bounds__(256) void link_encoder_kernel(
    const int*   __restrict__ n_id,
    const float* __restrict__ t_ref,
    const float* __restrict__ msg_store,
    const float* __restrict__ t_store,
    const int*   __restrict__ msg_count,
    const float* __restrict__ time_w,
    const float* __restrict__ time_b,
    const float* __restrict__ tok_ln_g,
    const float* __restrict__ tok_ln_b,
    const unsigned short* __restrict__ tokw_bf,
    const float* __restrict__ tok_b,
    const float* __restrict__ ch_ln_g,
    const float* __restrict__ ch_ln_b,
    const unsigned short* __restrict__ chw_bf,
    const float* __restrict__ ch_b,
    float* __restrict__ out,
    int batch) {
    extern __shared__ char smem_raw[];
    float*          Xf  = (float*)smem_raw;                        // [ROWS][LDF] f32
    unsigned short* Xbf = (unsigned short*)(Xf + ROWS * LDF);      // [ROWS][LDF] bf16
    float*          muv = (float*)(Xbf + ROWS * LDF);              // [ROWS]
    float*          rsv = muv + ROWS;                              // [ROWS]

    const int tid       = threadIdx.x;
    const int wave      = tid >> 5;
    const int lane      = tid & 31;
    const int item_base = blockIdx.x * ITEMS;

    // --- Phase A: irregular gather + temporal cosine encoding + mask ---
    for (int idx = tid; idx < ROWS * DIMS; idx += 256) {
        int r = idx >> 8;          // row 0..79
        int c = idx & 255;         // channel 0..255
        int item = r / SZ, s = r - item * SZ;
        int gitem = item_base + item;
        float val = 0.f;
        if (gitem < batch) {
            int node = n_id[gitem];
            int mc   = msg_count[node];
            if (s < mc) {
                if (c < TD) {
                    float dt = t_ref[gitem] - t_store[node * SZ + s];
                    val = cosf(dt * time_w[c] + time_b[c]);
                } else {
                    val = msg_store[(size_t)node * (SZ * HID) + (size_t)s * HID + (c - TD)];
                }
            }
        }
        Xf[r * LDF + c] = val;
    }
    __syncthreads();

    // --- Two channel-mixing blocks ---
    mixer_block(Xf, Xbf, muv, rsv, tok_ln_g, tok_ln_b, tokw_bf, tok_b, wave, lane, tid);
    mixer_block(Xf, Xbf, muv, rsv, ch_ln_g,  ch_ln_b,  chw_bf,  ch_b,  wave, lane, tid);

    // --- Mean over the SZ slots per item ---
    for (int idx = tid; idx < ITEMS * DIMS; idx += 256) {
        int item = idx >> 8, c = idx & 255;
        int gitem = item_base + item;
        if (gitem < batch) {
            float acc = 0.f;
            #pragma unroll
            for (int s = 0; s < SZ; ++s) acc += Xf[(item * SZ + s) * LDF + c];
            out[(size_t)gitem * DIMS + c] = acc * (1.0f / SZ);
        }
    }
}

// ---------------------------------------------------------------------------
extern "C" void kernel_launch(void* const* d_in, const int* in_sizes, int n_in,
                              void* d_out, int out_size, void* d_ws, size_t ws_size,
                              hipStream_t stream) {
    const int*   n_id      = (const int*)  d_in[0];
    const float* t_ref     = (const float*)d_in[1];
    const float* msg_store = (const float*)d_in[2];
    const float* t_store   = (const float*)d_in[3];
    const int*   msg_count = (const int*)  d_in[4];
    const float* time_w    = (const float*)d_in[5];
    const float* time_b    = (const float*)d_in[6];
    const float* tok_ln_g  = (const float*)d_in[7];
    const float* tok_ln_b  = (const float*)d_in[8];
    const float* tok_w     = (const float*)d_in[9];
    const float* tok_b     = (const float*)d_in[10];
    const float* ch_ln_g   = (const float*)d_in[11];
    const float* ch_ln_b   = (const float*)d_in[12];
    const float* ch_w      = (const float*)d_in[13];
    const float* ch_b      = (const float*)d_in[14];
    float* out = (float*)d_out;

    const int batch = in_sizes[0];
    const int nw    = DIMS * DIMS;

    unsigned short* tokw_bf = (unsigned short*)d_ws;
    unsigned short* chw_bf  = tokw_bf + nw;

    convert_weights_kernel<<<(nw + 255) / 256, 256, 0, stream>>>(
        tok_w, ch_w, tokw_bf, chw_bf, nw);

    const int nblocks = (batch + ITEMS - 1) / ITEMS;     // 5000
    const size_t smem = (size_t)ROWS * LDF * sizeof(float)
                      + (size_t)ROWS * LDF * sizeof(unsigned short)
                      + (size_t)2 * ROWS * sizeof(float);  // ~127.4 KB
    link_encoder_kernel<<<nblocks, 256, smem, stream>>>(
        n_id, t_ref, msg_store, t_store, msg_count, time_w, time_b,
        tok_ln_g, tok_ln_b, tokw_bf, tok_b,
        ch_ln_g, ch_ln_b, chw_bf, ch_b,
        out, batch);
}
